// HarmonicOscillator_90142773608850
// MI455X (gfx1250) — matrix operand order repaired and checked by reading, not compiled, not run on hardware
//
#include <hip/hip_runtime.h>
#include <hip/hip_bf16.h>
#include <math.h>

#define NH   48
#define CIN  256
#define LF   200
#define NB   4
#define SEGMENT 960
#define LW   (LF * SEGMENT)        // 192000
#define COLS (NB * LF)             // 800
#define SR_INV (1.0f / 48000.0f)
#define TWO_PI 6.283185307179586f

typedef __attribute__((ext_vector_type(16))) __bf16 v16bf;
typedef __attribute__((ext_vector_type(8)))  float  v8f;
typedef __attribute__((ext_vector_type(4)))  unsigned int v4u;
typedef __attribute__((ext_vector_type(8)))  int    v8i;
typedef __attribute__((ext_vector_type(4)))  int    v4i;

// ---------------------------------------------------------------------------
// Kernel 1: mag[n,h,l] = exp(min(sum_c x[n,c,l]*W[h,c] + bias[h], 4))
// GEMM M=48 x N=800 x K=256, bf16 WMMA, one 16x16 tile per wave.
// ---------------------------------------------------------------------------
__global__ void __launch_bounds__(256)
mag_gemm_wmma(const float* __restrict__ x, const float* __restrict__ W,
              const float* __restrict__ bias, float* __restrict__ mag)
{
    const int lane = threadIdx.x & 31;
    const int wave = threadIdx.x >> 5;
    const int tile = blockIdx.x * 8 + wave;
    const int MT = NH / 16;                 // 3
    const int NTILES = MT * (COLS / 16);    // 150
    if (tile >= NTILES) return;             // wave-uniform: EXEC all-1s for WMMA waves

    const int tileM = tile % MT;
    const int tileN = tile / MT;

    const int half = lane >> 4;             // 0: lanes 0-15, 1: lanes 16-31
    const int m    = lane & 15;
    const int h    = tileM * 16 + m;        // A row (harmonic index)
    const int col  = tileN * 16 + m;        // B column -> (n, l)
    const int nb   = col / LF;
    const int l    = col - nb * LF;

    v8f acc = {};
    #pragma unroll
    for (int kk = 0; kk < CIN; kk += 32) {
        v16bf a, bm;
        #pragma unroll
        for (int i = 0; i < 16; ++i) {
            // A 16x32 bf16: lanes 0-15 hold K=0..15, lanes 16-31 hold K=16..31
            int ka = kk + half * 16 + i;
            a[i] = (__bf16)W[h * CIN + ka];
            // B 32x16 bf16: lanes 0-15 hold K=0..15, lanes 16-31 hold K=16..31
            int kb = kk + half * 16 + i;
            bm[i] = (__bf16)x[(nb * CIN + kb) * LF + l];
        }
        acc = __builtin_amdgcn_wmma_f32_16x16x32_bf16(false, a, false, bm,
                                                      (short)0, acc, false, false);
    }

    // C/D f32 layout: VGPR r -> lanes 0-15: M=r, lanes 16-31: M=r+8; N = lane&15
    #pragma unroll
    for (int r = 0; r < 8; ++r) {
        int hr = tileM * 16 + r + half * 8;
        float v = __expf(fminf(acc[r] + bias[hr], 4.0f));
        mag[(nb * NH + hr) * LF + l] = v;
    }
}

// ---------------------------------------------------------------------------
// Kernel 2: phase[n][j] = inclusive cumsum_j( f0_w[n][j] / SR )
// One block per batch; chunked partial sums + LDS Hillis-Steele scan.
// ---------------------------------------------------------------------------
__global__ void __launch_bounds__(256)
phase_scan(const float* __restrict__ f0, float* __restrict__ phase)
{
    const int n = blockIdx.x;
    const int t = threadIdx.x;
    const int CH = LW / 256;                // 750
    const float* f = f0 + n * LF;
    __shared__ float s[256];

    auto f0w = [&](int j) -> float {
        float pos = fmaxf((j + 0.5f) * (1.0f / SEGMENT) - 0.5f, 0.0f);
        int i0 = min((int)floorf(pos), LF - 1);
        int i1 = min(i0 + 1, LF - 1);
        float w = pos - (float)i0;
        return f[i0] * (1.0f - w) + f[i1] * w;
    };

    const int j0 = t * CH;
    float local = 0.0f;
    for (int j = j0; j < j0 + CH; ++j)
        local += f0w(j) * SR_INV;
    s[t] = local;
    __syncthreads();

    #pragma unroll
    for (int off = 1; off < 256; off <<= 1) {
        float v = s[t];
        float add = (t >= off) ? s[t - off] : 0.0f;
        __syncthreads();
        s[t] = v + add;
        __syncthreads();
    }
    float run = (t == 0) ? 0.0f : s[t - 1];

    float* ph = phase + n * LW;
    for (int j = j0; j < j0 + CH; ++j) {
        run += f0w(j) * SR_INV;
        ph[j] = run;
    }
}

// ---------------------------------------------------------------------------
// Kernel 3: out[n,j] = (1/48) * sum_h sin(2*pi*(h+1)*phase) * mag_w[n,h,j]
// One block per (batch, segment). The 48x3 f32 mag-frame tile (rows strided
// by LF in memory) is DMA'd to LDS by the Tensor Data Mover (TENSORcnt),
// then every sample reads it as LDS broadcasts. Chebyshev recurrence:
// 1 sincos + 48 FMAs per sample instead of 48 sins.
// ---------------------------------------------------------------------------
__global__ void __launch_bounds__(320)
synth_tdm(const float* __restrict__ mag, const float* __restrict__ phase,
          float* __restrict__ out)
{
    __shared__ float smem[NH * 3];          // 48 harmonics x 3 frames

    const int n   = blockIdx.x / LF;        // batch
    const int seg = blockIdx.x - n * LF;    // frame segment (960 samples)
    const int base = min(max(seg - 1, 0), LF - 3);   // first staged frame

    if (threadIdx.x < 32) {                 // wave 0 drives the TDM
        unsigned long long gaddr =
            (unsigned long long)(const void*)(mag + (n * NH) * LF + base);
        unsigned lds_off = (unsigned)(unsigned long long)(const void*)&smem[0];

        // D# group 0: count=1 | lds_addr | global_addr[56:0] | type=2
        v4u g0;
        g0[0] = 1u;
        g0[1] = lds_off;
        g0[2] = (unsigned)(gaddr & 0xFFFFFFFFull);
        g0[3] = (unsigned)((gaddr >> 32) & 0x1FFFFFFull) | (2u << 30);

        // D# group 1: data_size=4B; tensor 200x48, stride0=200; tile 3x48
        v8i g1;
        g1[0] = (2 << 16);                          // data_size = 4 bytes
        g1[1] = (int)((LF & 0xFFFF) << 16);         // tensor_dim0[15:0]
        g1[2] = (int)((LF >> 16) | ((NH & 0xFFFF) << 16));  // dim0 hi | dim1 lo
        g1[3] = (int)((NH >> 16) | (3 << 16));      // dim1 hi | tile_dim0 = 3
        g1[4] = NH;                                 // tile_dim1 = 48, tile_dim2 = 0
        g1[5] = LF;                                 // tensor_dim0_stride[31:0]
        g1[6] = 0;                                  // stride0 hi | stride1 lo
        g1[7] = 0;                                  // stride1 hi

        v4i g2 = {};                                // 2D tensor: groups 2/3 unused
        v4i g3 = {};
        v8i g4 = {};                                // extra group (clang-23 6-arg form)

        __builtin_amdgcn_tensor_load_to_lds(g0, g1, g2, g3, g4, 0);
        __builtin_amdgcn_s_wait_tensorcnt(0);
    }
    __syncthreads();

    const float* ph = phase + n * LW + seg * SEGMENT;
    float*       po = out   + n * LW + seg * SEGMENT;

    #pragma unroll
    for (int k = 0; k < 3; ++k) {
        const int jj = threadIdx.x + k * 320;       // sample within segment
        const int j  = seg * SEGMENT + jj;          // global sample index

        float p   = ph[jj];
        float pos = fmaxf((j + 0.5f) * (1.0f / SEGMENT) - 0.5f, 0.0f);
        int i0 = min((int)floorf(pos), LF - 1);
        int i1 = min(i0 + 1, LF - 1);
        float w  = pos - (float)i0;
        int  fi0 = i0 - base;                       // in [0,2] by construction
        int  fi1 = i1 - base;

        float theta = TWO_PI * p;
        float s1, c1;
        __sincosf(theta, &s1, &c1);
        float c2   = 2.0f * c1;
        float skm1 = 0.0f;                          // sin(0*theta)
        float sk   = s1;                            // sin(1*theta)

        const float* m0 = smem + fi0;               // stride 3 per harmonic
        const float* m1 = smem + fi1;
        float accum = 0.0f;
        #pragma unroll
        for (int hh = 0; hh < NH; ++hh) {
            float mg = m0[hh * 3] * (1.0f - w) + m1[hh * 3] * w;
            accum = fmaf(sk, mg, accum);
            float snext = fmaf(c2, sk, -skm1);
            skm1 = sk;
            sk = snext;
        }
        po[jj] = accum * (1.0f / NH);
    }
}

// ---------------------------------------------------------------------------
extern "C" void kernel_launch(void* const* d_in, const int* in_sizes, int n_in,
                              void* d_out, int out_size, void* d_ws, size_t ws_size,
                              hipStream_t stream)
{
    (void)in_sizes; (void)n_in; (void)out_size; (void)ws_size;
    const float* x    = (const float*)d_in[0];   // (4, 256, 200)
    const float* f0   = (const float*)d_in[1];   // (4, 1, 200)
    const float* W    = (const float*)d_in[2];   // (48, 256)
    const float* bias = (const float*)d_in[3];   // (48,)
    float* out = (float*)d_out;                  // (4, 1, 192000)

    float* mag   = (float*)d_ws;                 // NB*NH*LF  = 38400 floats
    float* phase = mag + NB * NH * LF;           // NB*LW     = 768000 floats

    const int ntiles = (NH / 16) * (COLS / 16);  // 150
    mag_gemm_wmma<<<(ntiles + 7) / 8, 256, 0, stream>>>(x, W, bias, mag);
    phase_scan<<<NB, 256, 0, stream>>>(f0, phase);
    synth_tdm<<<NB * LF, 320, 0, stream>>>(mag, phase, out);
}
